// MultiHeadAttention_75428215652432
// MI455X (gfx1250) — matrix operand order repaired and checked by reading, not compiled, not run on hardware
//
#include <hip/hip_runtime.h>

typedef __bf16 bf16;
typedef __attribute__((ext_vector_type(16))) __bf16 v16bf;
typedef __attribute__((ext_vector_type(8)))  __bf16 bf16x8;
typedef __attribute__((ext_vector_type(8)))  float  v8f;

#define DD   1024   // embed dim
#define SS   2048   // seq len
#define BB   4      // batch

// ---------------------------------------------------------------------------
// Fragment loaders (wave32, v_wmma_f32_16x16x32_bf16)
// A 16x32 (MxK): lane<16 -> row=lane, elems K = {h*8+0..7, 16+h*8+0..7}
// B 32x16 (KxN): lane -> col = lane&15; elems K = {h*8+0..7, 16+h*8+0..7}
//                (source layout: row index == N column, contiguous in K)
// C/D 16x16:     lane l, vgpr j -> (m = j + 8*(l>>4), n = l&15)
// ---------------------------------------------------------------------------
__device__ __forceinline__ v16bf load_fragA(const bf16* __restrict__ base,
                                            int stride, int row0, int k0, int lane) {
  const int h = (lane >> 4) & 1;
  const int r = lane & 15;
  const bf16* p = base + (size_t)(row0 + r) * stride + k0 + h * 8;
  bf16x8 lo = *(const bf16x8*)(p);       // K = h*8 + 0..7
  bf16x8 hi = *(const bf16x8*)(p + 16);  // K = 16 + h*8 + 0..7
  v16bf f;
#pragma unroll
  for (int i = 0; i < 8; ++i) { f[i] = lo[i]; f[i + 8] = hi[i]; }
  return f;
}

__device__ __forceinline__ v16bf load_fragB(const bf16* __restrict__ base,
                                            int stride, int col0, int k0, int lane) {
  const int h = (lane >> 4) & 1;
  const int r = lane & 15;
  const bf16* p = base + (size_t)(col0 + r) * stride + k0 + h * 8;
  bf16x8 lo = *(const bf16x8*)(p);
  bf16x8 hi = *(const bf16x8*)(p + 16);
  v16bf f;
#pragma unroll
  for (int i = 0; i < 8; ++i) { f[i] = lo[i]; f[i + 8] = hi[i]; }
  return f;
}

__device__ __forceinline__ v8f wmma_bf16(v16bf a, v16bf b, v8f c) {
  return __builtin_amdgcn_wmma_f32_16x16x32_bf16(false, a, false, b, (short)0, c,
                                                 false, false);
}

// 32x64-per-wave macro-tiled bf16 GEMM core: 8 WMMAs per 6 fragment loads.
// (2x4 was chosen over 4x4: 4x4 spilled accumulators to scratch.)
#define MT 2
#define NT 4
__device__ __forceinline__ void gemm_tile(const bf16* __restrict__ A, int lda,
                                          const bf16* __restrict__ B, int ldb,
                                          int row0, int col0, int Kdim, int lane,
                                          v8f acc[MT][NT]) {
  for (int kc = 0; kc < Kdim; kc += 32) {
    v16bf af[MT];
    v16bf bfr[NT];
#pragma unroll
    for (int mi = 0; mi < MT; ++mi) af[mi] = load_fragA(A, lda, row0 + mi * 16, kc, lane);
#pragma unroll
    for (int ni = 0; ni < NT; ++ni) bfr[ni] = load_fragB(B, ldb, col0 + ni * 16, kc, lane);
#pragma unroll
    for (int mi = 0; mi < MT; ++mi)
#pragma unroll
      for (int ni = 0; ni < NT; ++ni)
        acc[mi][ni] = wmma_bf16(af[mi], bfr[ni], acc[mi][ni]);
  }
}

#define ACC_ZERO(acc)                                                          \
  _Pragma("unroll") for (int mi = 0; mi < MT; ++mi)                            \
  _Pragma("unroll") for (int ni = 0; ni < NT; ++ni)                            \
      acc[mi][ni] = (v8f){0.f, 0.f, 0.f, 0.f, 0.f, 0.f, 0.f, 0.f};

// ---------------------------------------------------------------------------
// K0a: xb = bf16(x)          (8192*1024 elems)
// ---------------------------------------------------------------------------
__global__ void xb_kernel(const float* __restrict__ x, bf16* __restrict__ xb) {
  size_t idx = (size_t)blockIdx.x * 256 + threadIdx.x;
  xb[idx] = (bf16)x[idx];
}

// ---------------------------------------------------------------------------
// K0b: Wt[mat][n][k] = bf16(W[k][n])   (3 x 1024 x 1024)
// ---------------------------------------------------------------------------
__global__ void wt_kernel(const float* __restrict__ Wq, const float* __restrict__ Wk,
                          const float* __restrict__ Wv, bf16* __restrict__ Wt) {
  size_t idx = (size_t)blockIdx.x * 256 + threadIdx.x;  // [mat][n][k]
  int k   = (int)(idx & (DD - 1));
  int n   = (int)((idx >> 10) & (DD - 1));
  int mat = (int)(idx >> 20);
  const float* W = (mat == 0) ? Wq : ((mat == 1) ? Wk : Wv);
  Wt[idx] = (bf16)W[(size_t)k * DD + n];
}

// ---------------------------------------------------------------------------
// K1: Q/K/V projections, 32x64 per wave. Q,K row-major bf16 [B*S, D];
//     V stored transposed per batch: Vt[b][d][s].
// ---------------------------------------------------------------------------
__global__ void __launch_bounds__(256, 1)
qkv_kernel(const bf16* __restrict__ xb, const bf16* __restrict__ Wt,
           const float* __restrict__ bq, const float* __restrict__ bk,
           const float* __restrict__ bv,
           bf16* __restrict__ Qb, bf16* __restrict__ Kb, bf16* __restrict__ Vt) {
  const int lane = threadIdx.x & 31;
  const int wid  = blockIdx.x * (blockDim.x >> 5) + (threadIdx.x >> 5);
  const int nt   = wid & 15;           // 16 N blocks of 64
  const int mt   = (wid >> 4) & 255;   // 256 M blocks of 32
  const int mat  = wid >> 12;          // 3 matrices
  if (mat >= 3) return;

  const bf16* Wm = Wt + (size_t)mat * DD * DD;
  const int row0 = mt * 32, col0 = nt * 64;

  v8f acc[MT][NT];
  ACC_ZERO(acc);
  gemm_tile(xb, DD, Wm, DD, row0, col0, DD, lane, acc);

  const float* bias = (mat == 0) ? bq : ((mat == 1) ? bk : bv);
  const int h = lane >> 4, r = lane & 15;

  if (mat == 2) {                       // V -> transposed store Vt[b][d][s]
    const int bidx  = row0 >> 11;       // row0 / SS (32-row block within one batch)
    const int srow0 = row0 & (SS - 1);
    bf16* vt = Vt + (size_t)bidx * DD * SS;
#pragma unroll
    for (int mi = 0; mi < MT; ++mi)
#pragma unroll
      for (int ni = 0; ni < NT; ++ni) {
        const float bn = bias[col0 + ni * 16 + r];
#pragma unroll
        for (int j = 0; j < 8; ++j)
          vt[(size_t)(col0 + ni * 16 + r) * SS + srow0 + mi * 16 + j + 8 * h] =
              (bf16)(acc[mi][ni][j] + bn);
      }
  } else {
    bf16* out = (mat == 0) ? Qb : Kb;
#pragma unroll
    for (int mi = 0; mi < MT; ++mi)
#pragma unroll
      for (int ni = 0; ni < NT; ++ni) {
        const float bn = bias[col0 + ni * 16 + r];
#pragma unroll
        for (int j = 0; j < 8; ++j)
          out[(size_t)(row0 + mi * 16 + j + 8 * h) * DD + col0 + ni * 16 + r] =
              (bf16)(acc[mi][ni][j] + bn);
      }
  }
}

// ---------------------------------------------------------------------------
// K2: S[b][q][k] = (Q . K) / sqrt(128), bf16 out, 32x64 per wave
// ---------------------------------------------------------------------------
__global__ void __launch_bounds__(256, 1)
scores_kernel(const bf16* __restrict__ Qb, const bf16* __restrict__ Kb,
              bf16* __restrict__ Sb) {
  const int lane = threadIdx.x & 31;
  const int wid  = blockIdx.x * (blockDim.x >> 5) + (threadIdx.x >> 5);
  const int nt   = wid & 31;           // 32 key blocks of 64
  const int mt   = (wid >> 5) & 63;    // 64 query blocks of 32
  const int b    = wid >> 11;
  if (b >= BB) return;

  const bf16* Q = Qb + (size_t)b * SS * DD;
  const bf16* K = Kb + (size_t)b * SS * DD;
  const int row0 = mt * 32, col0 = nt * 64;

  v8f acc[MT][NT];
  ACC_ZERO(acc);
  gemm_tile(Q, DD, K, DD, row0, col0, DD, lane, acc);  // K row-major == K^T col-major

  const float scale = 0.0883883476483184f;  // 1/sqrt(HEAD_DIM=128)
  bf16* Srow = Sb + (size_t)b * SS * SS;
  const int h = lane >> 4, r = lane & 15;
#pragma unroll
  for (int mi = 0; mi < MT; ++mi)
#pragma unroll
    for (int ni = 0; ni < NT; ++ni)
#pragma unroll
      for (int j = 0; j < 8; ++j)
        Srow[(size_t)(row0 + mi * 16 + j + 8 * h) * SS + col0 + ni * 16 + r] =
            (bf16)(acc[mi][ni][j] * scale);
}

// ---------------------------------------------------------------------------
// K3: in-place row softmax over 2048 cols; one 256-thread block per row.
// ---------------------------------------------------------------------------
__global__ void softmax_kernel(bf16* __restrict__ Sb) {
  __shared__ float red[256];
  bf16* p = Sb + (size_t)blockIdx.x * SS;
  const int t = threadIdx.x;

  float v[8];
  float m = -3.0e38f;
#pragma unroll
  for (int i = 0; i < 8; ++i) {
    v[i] = (float)p[t + i * 256];
    m = fmaxf(m, v[i]);
  }
  red[t] = m; __syncthreads();
  for (int s2 = 128; s2 > 0; s2 >>= 1) {
    if (t < s2) red[t] = fmaxf(red[t], red[t + s2]);
    __syncthreads();
  }
  m = red[0]; __syncthreads();

  float sum = 0.f;
#pragma unroll
  for (int i = 0; i < 8; ++i) { v[i] = __expf(v[i] - m); sum += v[i]; }
  red[t] = sum; __syncthreads();
  for (int s2 = 128; s2 > 0; s2 >>= 1) {
    if (t < s2) red[t] += red[t + s2];
    __syncthreads();
  }
  const float inv = 1.0f / red[0];
#pragma unroll
  for (int i = 0; i < 8; ++i) p[t + i * 256] = (bf16)(v[i] * inv);
}

// ---------------------------------------------------------------------------
// K4: O[b][q][d] = P[b][q][:] . V[b][:][d]  (Vt is [b][d][s]), f32 out,
//     32x64 per wave
// ---------------------------------------------------------------------------
__global__ void __launch_bounds__(256, 1)
out_kernel(const bf16* __restrict__ Pb, const bf16* __restrict__ Vt,
           float* __restrict__ out) {
  const int lane = threadIdx.x & 31;
  const int wid  = blockIdx.x * (blockDim.x >> 5) + (threadIdx.x >> 5);
  const int nt   = wid & 15;           // 16 d blocks of 64
  const int mt   = (wid >> 4) & 63;    // 64 q blocks of 32
  const int b    = wid >> 10;
  if (b >= BB) return;

  const bf16* P = Pb + (size_t)b * SS * SS;
  const bf16* V = Vt + (size_t)b * DD * SS;
  const int row0 = mt * 32, col0 = nt * 64;

  v8f acc[MT][NT];
  ACC_ZERO(acc);
  gemm_tile(P, SS, V, SS, row0, col0, SS, lane, acc);  // Vt row d = B column d

  float* o = out + (size_t)b * SS * DD;
  const int h = lane >> 4, r = lane & 15;
#pragma unroll
  for (int mi = 0; mi < MT; ++mi)
#pragma unroll
    for (int ni = 0; ni < NT; ++ni)
#pragma unroll
      for (int j = 0; j < 8; ++j)
        o[(size_t)(row0 + mi * 16 + j + 8 * h) * DD + col0 + ni * 16 + r] =
            acc[mi][ni][j];
}

// ---------------------------------------------------------------------------
extern "C" void kernel_launch(void* const* d_in, const int* in_sizes, int n_in,
                              void* d_out, int out_size, void* d_ws, size_t ws_size,
                              hipStream_t stream) {
  const float* x  = (const float*)d_in[0];
  const float* Wq = (const float*)d_in[1];
  const float* bq = (const float*)d_in[2];
  const float* Wk = (const float*)d_in[3];
  const float* bk = (const float*)d_in[4];
  const float* Wv = (const float*)d_in[5];
  const float* bv = (const float*)d_in[6];
  float* out = (float*)d_out;

  char* ws = (char*)d_ws;
  const size_t MB = 1024 * 1024;
  // workspace layout (bytes)
  bf16* Xb = (bf16*)(ws);              // 16 MB: bf16(x), 8192*1024
  bf16* Wt = (bf16*)(ws + 16 * MB);    //  6 MB: 3*1024*1024 (transposed weights)
  bf16* Qb = (bf16*)(ws + 22 * MB);    // 16 MB
  bf16* Kb = (bf16*)(ws + 38 * MB);    // 16 MB
  bf16* Vt = (bf16*)(ws + 54 * MB);    // 16 MB (transposed V)
  bf16* Sb = (bf16*)(ws + 70 * MB);    // 32 MB: 4*2048*2048

  // K0a: x -> bf16
  xb_kernel<<<32768, 256, 0, stream>>>(x, Xb);
  // K0b: weight transpose + bf16 convert
  wt_kernel<<<12288, 256, 0, stream>>>(Wq, Wk, Wv, Wt);

  // K1: QKV projections: 3 * 256 * 16 waves / 8 waves per block
  qkv_kernel<<<1536, 256, 0, stream>>>(Xb, Wt, bq, bk, bv, Qb, Kb, Vt);

  // K2: scores: 4 * 64 * 32 waves / 8
  scores_kernel<<<1024, 256, 0, stream>>>(Qb, Kb, Sb);

  // K3: softmax: one block per row (4*2048 rows)
  softmax_kernel<<<8192, 256, 0, stream>>>(Sb);

  // K4: output GEMM: 4 * 64 * 16 waves / 8
  out_kernel<<<512, 256, 0, stream>>>(Sb, Vt, out);
}
